// SAM3_23287312679297
// MI455X (gfx1250) — matrix operand order repaired and checked by reading, not compiled, not run on hardware
//
#include <hip/hip_runtime.h>
#include <hip/hip_bf16.h>
#include <math.h>

typedef __attribute__((ext_vector_type(16))) _Float16 v16h;
typedef __attribute__((ext_vector_type(8)))  float    v8f;

constexpr int Bv = 4, Qv = 200, HEADSv = 8, Dv = 256, Lv = 6, Tv = 32;
constexpr int INTERv = 2048, HWv = 4096, HDv = 32, NQ1 = 201;
constexpr int SSP = 224;  // self-attn score row stride, 201 padded to %32==0
constexpr float ATTN_SCALE = 0.17677669529663689f;  // 1/sqrt(32)

// ---------------------------------------------------------------------------
// WMMA GEMM: C[M,N] (+= / =) scale * A[M,K] x B[K,N]  (+bias, +relu)
// A, B f32 in memory, packed to f16 in registers (v_cvt_pk_rtz_f16_f32),
// f32 accumulate via v_wmma_f32_16x16x32_f16. One 16x16 tile per wave32.
// Requires K % 32 == 0 (callers pad K). Branch-free loads: row/col clamped
// to valid range; out-of-tile lanes compute garbage that is never stored.
// TRANS_B: B given as BT[N,K] row-major (contiguous K pairs -> float2 loads).
// Batched over (batch x heads) with independent strides.
// ---------------------------------------------------------------------------
struct GemmArgs {
  const float* A; const float* B; float* C; const float* bias;
  int M, N, K;
  int lda, ldb, ldc;
  long long sAb, sBb, sCb;
  long long sAh, sBh, sCh;
  int batch, heads;
  float scale;
  int add_c, relu;
};

template <bool TRANS_B>
__global__ void __launch_bounds__(256) k_gemm_wmma(GemmArgs g) {
  const int lane = threadIdx.x & 31;
  const long long wave = (long long)blockIdx.x * (blockDim.x >> 5) + (threadIdx.x >> 5);
  const int tm = (g.M + 15) >> 4, tn = (g.N + 15) >> 4;
  const int tiles = tm * tn;
  const long long total = (long long)g.batch * g.heads * tiles;
  if (wave >= total) return;                 // wave-uniform: EXEC stays full for WMMA
  const int t  = (int)(wave % tiles);
  const int bh = (int)(wave / tiles);
  const int hh = bh % g.heads, bb = bh / g.heads;
  const int tile_n = t % tn, tile_m = t / tn;
  const float* A = g.A + bb * g.sAb + hh * g.sAh;
  const float* B = g.B + bb * g.sBb + hh * g.sBh;
  float*       C = g.C + bb * g.sCb + hh * g.sCh;

  const int ml  = lane & 15;
  const int khf = lane >> 4;
  const int row  = tile_m * 16 + ml;                // A row handled by this lane
  const int coln = tile_n * 16 + ml;                // B column handled by this lane
  const int rowc = (row  < g.M) ? row  : g.M - 1;   // clamp: garbage never stored
  const int colc = (coln < g.N) ? coln : g.N - 1;
  const float* Arow = A + (long long)rowc * g.lda;
  const float* Bcol = B + (long long)colc * g.ldb;  // used when TRANS_B

  v8f acc = {};
  for (int k0 = 0; k0 < g.K; k0 += 32) {
    __builtin_prefetch((const void*)(Arow + k0 + 32), 0, 1);  // global_prefetch_b8
    v16h af, bf;
    // A fragment (16-bit A 16x32: lanes 0-15 K{0..7,16..23}, lanes 16-31 K{8..15,24..31})
#pragma unroll
    for (int r = 0; r < 8; ++r) {
      int kk = k0 + ((r < 4) ? (khf * 8 + 2 * r) : (16 + khf * 8 + 2 * (r - 4)));
      float2 v2 = *(const float2*)(Arow + kk);
      auto p = __builtin_amdgcn_cvt_pkrtz(v2.x, v2.y);
      af[2 * r] = (_Float16)p[0]; af[2 * r + 1] = (_Float16)p[1];
    }
    // B fragment (lanes 0-15: N=lane, K{0..7,16..23}; lanes 16-31: K{8..15,24..31})
#pragma unroll
    for (int r = 0; r < 8; ++r) {
      int kk = k0 + ((r < 4) ? (khf * 8 + 2 * r) : (16 + khf * 8 + 2 * (r - 4)));
      float x0, x1;
      if (TRANS_B) {
        float2 v2 = *(const float2*)(Bcol + kk);
        x0 = v2.x; x1 = v2.y;
      } else {
        x0 = B[(long long)kk * g.ldb + colc];
        x1 = B[(long long)(kk + 1) * g.ldb + colc];
      }
      auto p = __builtin_amdgcn_cvt_pkrtz(x0, x1);
      bf[2 * r] = (_Float16)p[0]; bf[2 * r + 1] = (_Float16)p[1];
    }
    acc = __builtin_amdgcn_wmma_f32_16x16x32_f16(false, af, false, bf, (short)0, acc,
                                                 false, false);
  }
  // Epilogue: C/D layout — lane holds column coln, VGPR r -> M = r + 8*khf
#pragma unroll
  for (int r = 0; r < 8; ++r) {
    int m = tile_m * 16 + r + 8 * khf;
    if (m < g.M && coln < g.N) {
      long long off = (long long)m * g.ldc + coln;
      float v = acc[r] * g.scale;
      if (g.add_c) v += C[off];
      if (g.bias)  v += g.bias[coln];
      if (g.relu)  v = fmaxf(v, 0.f);
      C[off] = v;
    }
  }
}

// ---------------------------------------------------------------------------
// LayerNorm over 256 columns, one wave per row; optional residual (in+res).
// ---------------------------------------------------------------------------
__global__ void __launch_bounds__(256) k_ln(const float* __restrict__ in,
                                            const float* __restrict__ res,
                                            const float* __restrict__ gg,
                                            const float* __restrict__ bb,
                                            float* __restrict__ out, int rows) {
  int lane = threadIdx.x & 31;
  int row = blockIdx.x * (blockDim.x >> 5) + (threadIdx.x >> 5);
  if (row >= rows) return;
  const float* p = in + (long long)row * 256;
  const float* r = res ? res + (long long)row * 256 : nullptr;
  float v[8]; float s = 0.f;
#pragma unroll
  for (int i = 0; i < 8; ++i) {
    float x = p[lane + 32 * i];
    if (r) x += r[lane + 32 * i];
    v[i] = x; s += x;
  }
  for (int o = 16; o; o >>= 1) s += __shfl_xor(s, o, 32);
  float mean = s * (1.f / 256.f);
  float var = 0.f;
#pragma unroll
  for (int i = 0; i < 8; ++i) { float d = v[i] - mean; var += d * d; }
  for (int o = 16; o; o >>= 1) var += __shfl_xor(var, o, 32);
  float rstd = rsqrtf(var * (1.f / 256.f) + 1e-6f);
  float* q = out + (long long)row * 256;
#pragma unroll
  for (int i = 0; i < 8; ++i) {
    int c = lane + 32 * i;
    q[c] = (v[i] - mean) * rstd * gg[c] + bb[c];
  }
}

// ---------------------------------------------------------------------------
// Softmax over `cols` (row stride `ld`), one block per row, in place;
// optional key mask [b, cols].
// ---------------------------------------------------------------------------
__global__ void __launch_bounds__(256) k_softmax(float* __restrict__ io,
                                                 const int* __restrict__ mask,
                                                 int cols, int ld, int rows_per_b) {
  long long row = blockIdx.x;
  float* p = io + row * ld;
  int b = (int)(row / rows_per_b);
  __shared__ float smax[8];
  __shared__ float ssum[8];
  int tid = threadIdx.x, lane = tid & 31, wv = tid >> 5;
  float mx = -3.0e38f;
  for (int c = tid; c < cols; c += 256) {
    float v = p[c];
    if (mask && mask[b * cols + c] == 0) { v = -1.0e9f; p[c] = v; }
    mx = fmaxf(mx, v);
  }
  for (int o = 16; o; o >>= 1) mx = fmaxf(mx, __shfl_xor(mx, o, 32));
  if (lane == 0) smax[wv] = mx;
  __syncthreads();
  if (wv == 0) {
    float m2 = (lane < 8) ? smax[lane] : -3.0e38f;
    for (int o = 16; o; o >>= 1) m2 = fmaxf(m2, __shfl_xor(m2, o, 32));
    if (lane == 0) smax[0] = m2;
  }
  __syncthreads();
  mx = smax[0];
  float sum = 0.f;
  for (int c = tid; c < cols; c += 256) { float e = expf(p[c] - mx); p[c] = e; sum += e; }
  for (int o = 16; o; o >>= 1) sum += __shfl_xor(sum, o, 32);
  if (lane == 0) ssum[wv] = sum;
  __syncthreads();
  if (wv == 0) {
    float s2 = (lane < 8) ? ssum[lane] : 0.f;
    for (int o = 16; o; o >>= 1) s2 += __shfl_xor(s2, o, 32);
    if (lane == 0) ssum[0] = s2;
  }
  __syncthreads();
  float inv = 1.f / ssum[0];
  for (int c = tid; c < cols; c += 256) p[c] *= inv;
}

// ---------------------------------------------------------------------------
// Elementwise / glue kernels
// ---------------------------------------------------------------------------
__global__ void k_zero(float* o, int n) {
  int i = blockIdx.x * blockDim.x + threadIdx.x;
  if (i < n) o[i] = 0.f;
}
__global__ void k_add(float* o, const float* a, const float* b, int n) {
  int i = blockIdx.x * blockDim.x + threadIdx.x;
  if (i < n) o[i] = a[i] + b[i];
}
__global__ void k_add_bcast(float* o, const float* a, const float* b, int n, int bmod) {
  int i = blockIdx.x * blockDim.x + threadIdx.x;
  if (i < n) o[i] = a[i] + b[i % bmod];
}
__global__ void k_copy(const float* a, float* o, int n) {
  int i = blockIdx.x * blockDim.x + threadIdx.x;
  if (i < n) o[i] = a[i];
}
__global__ void k_slice(const float* h, float* out, int batch, int take, int start,
                        int per_b, int cols) {
  int i = blockIdx.x * blockDim.x + threadIdx.x;
  int n = batch * take * cols;
  if (i >= n) return;
  int d = i % cols, r = (i / cols) % take, b = i / (cols * take);
  out[i] = h[((long long)b * per_b + start + r) * cols + d];
}
__global__ void k_init_h(float* h, const float* pres_tok, const float* qe) {
  int i = blockIdx.x * blockDim.x + threadIdx.x;
  if (i >= Bv * NQ1 * Dv) return;
  int d = i % Dv, r = (i / Dv) % NQ1;
  h[i] = (r == 0) ? pres_tok[d] : qe[(r - 1) * Dv + d];
}
__global__ void k_init_ref(float* ref, const float* rp) {
  int i = blockIdx.x * blockDim.x + threadIdx.x;
  if (i >= Bv * Qv * 4) return;
  int qc = i % (Qv * 4);
  ref[i] = 1.f / (1.f + expf(-rp[qc]));
}
__global__ void k_qpos(const float* qpr, float* qpos) {
  int i = blockIdx.x * blockDim.x + threadIdx.x;
  if (i >= Bv * NQ1 * Dv) return;
  int d = i % Dv, r = (i / Dv) % NQ1, b = i / (Dv * NQ1);
  qpos[i] = (r == 0) ? 0.f : qpr[((long long)(b * Qv + (r - 1))) * Dv + d];
}
__global__ void k_sine_ref(const float* ref, float* out) {
  int i = blockIdx.x * blockDim.x + threadIdx.x;
  if (i >= Bv * Qv * 512) return;
  int d = i % 512, q = (i / 512) % Qv, b = i / (512 * Qv);
  int c = d >> 7, j = d & 127, fi = j >> 1;
  float x = ref[((b * Qv) + q) * 4 + c];
  float pos = x * 6.283185307179586f * powf(10000.f, -(float)fi * (1.f / 64.f));
  out[i] = (j & 1) ? cosf(pos) : sinf(pos);
}
__global__ void k_sine_rpb(const float* ref, float* out, int axis) {
  int i = blockIdx.x * blockDim.x + threadIdx.x;
  if (i >= Bv * Qv * 64 * 256) return;
  int d = i % 256, w = (i / 256) & 63, q = (i / (256 * 64)) % Qv, b = i / (256 * 64 * Qv);
  int c = d >> 7, j = d & 127, fi = j >> 1;
  const float* bx = ref + ((b * Qv) + q) * 4;
  float ctr = axis ? bx[1] : bx[0];
  float ext = axis ? bx[3] : bx[2];
  float edge = (c == 0) ? (ctr - 0.5f * ext) : (ctr + 0.5f * ext);
  float coord = ((float)w + 0.5f) * (1.f / 64.f);
  float pos = (coord - edge) * 6.283185307179586f * powf(10000.f, -(float)fi * (1.f / 64.f));
  out[i] = (j & 1) ? cosf(pos) : sinf(pos);
}
__global__ void k_bias_expand(const float* ex, const float* ey, float* vb) {
  int i = blockIdx.x * blockDim.x + threadIdx.x;
  if (i >= Bv * HEADSv * NQ1 * HWv) return;
  int col = i & (HWv - 1);
  int q1 = (i / HWv) % NQ1;
  int hd = (i / (HWv * NQ1)) % HEADSv;
  int b  = i / (HWv * NQ1 * HEADSv);
  float v = 0.f;
  if (q1 > 0) {
    int q = q1 - 1, y = col >> 6, x = col & 63;
    v = ey[(((b * Qv) + q) * 64 + y) * HEADSv + hd] +
        ex[(((b * Qv) + q) * 64 + x) * HEADSv + hd];
  }
  vb[i] = v;
}
__global__ void k_update_ref(float* ref, const float* delta, int n) {
  int i = blockIdx.x * blockDim.x + threadIdx.x;
  if (i >= n) return;
  float p = fminf(1.f - 1e-5f, fmaxf(1e-5f, ref[i]));
  float lg = logf(p / (1.f - p)) + delta[i];
  ref[i] = 1.f / (1.f + expf(-lg));
}
__global__ void k_boxes(const float* ref, float* out, int n) {
  int i = blockIdx.x * blockDim.x + threadIdx.x;
  if (i >= n) return;
  const float* r = ref + (long long)i * 4;
  float* o = out + (long long)i * 4;
  float cx = r[0], cy = r[1], w = r[2], h = r[3];
  o[0] = cx - 0.5f * w; o[1] = cy - 0.5f * h; o[2] = cx + 0.5f * w; o[3] = cy + 0.5f * h;
}
__global__ void k_pres(const float* praw, float* out) {
  int b = threadIdx.x;
  if (b < Bv) out[b] = fminf(10.f, fmaxf(-10.f, praw[b]));
}
__global__ void k_pool(const float* tf, const int* m, float* pooled) {
  int i = blockIdx.x * blockDim.x + threadIdx.x;
  if (i >= Bv * Dv) return;
  int b = i / Dv, d = i % Dv;
  float s = 0.f, cm = 0.f;
  for (int t = 0; t < Tv; ++t) {
    float mm = m[b * Tv + t] ? 1.f : 0.f;
    s += tf[((long long)(b * Tv + t)) * Dv + d] * mm;
    cm += mm;
  }
  pooled[i] = s / (cm + 1e-8f);
}
__global__ void k_logits(const float* qp, const float* tv, float* out) {
  int lane = threadIdx.x & 31;
  int wave = blockIdx.x * (blockDim.x >> 5) + (threadIdx.x >> 5);
  if (wave >= Bv * Qv) return;
  int b = wave / Qv;
  const float* q = qp + (long long)wave * Dv;
  const float* t = tv + (long long)b * Dv;
  float s = 0.f;
#pragma unroll
  for (int i = 0; i < 8; ++i) { int c = lane + 32 * i; s += q[c] * t[c]; }
  for (int o = 16; o; o >>= 1) s += __shfl_xor(s, o, 32);
  if (lane == 0) out[wave] = fminf(12.f, fmaxf(-12.f, s * 0.0625f));
}

// ---------------------------------------------------------------------------
// Host orchestration
// ---------------------------------------------------------------------------
static void launch_gemm(hipStream_t s, const float* A, const float* B, float* C,
                        const float* bias, int M, int N, int K, int lda, int ldb, int ldc,
                        long long sAb, long long sBb, long long sCb,
                        long long sAh, long long sBh, long long sCh,
                        int batch, int heads, float scale, bool transB, bool addC, bool relu) {
  GemmArgs g;
  g.A = A; g.B = B; g.C = C; g.bias = bias;
  g.M = M; g.N = N; g.K = K; g.lda = lda; g.ldb = ldb; g.ldc = ldc;
  g.sAb = sAb; g.sBb = sBb; g.sCb = sCb; g.sAh = sAh; g.sBh = sBh; g.sCh = sCh;
  g.batch = batch; g.heads = heads; g.scale = scale;
  g.add_c = addC ? 1 : 0; g.relu = relu ? 1 : 0;
  long long tiles = (long long)((M + 15) / 16) * ((N + 15) / 16) * batch * heads;
  int blocks = (int)((tiles + 7) / 8);
  if (transB) k_gemm_wmma<true ><<<blocks, 256, 0, s>>>(g);
  else        k_gemm_wmma<false><<<blocks, 256, 0, s>>>(g);
}

#define LAUNCH(kern, n, ...) kern<<<(((n) + 255) / 256), 256, 0, stream>>>(__VA_ARGS__)

extern "C" void kernel_launch(void* const* d_in, const int* in_sizes, int n_in,
                              void* d_out, int out_size, void* d_ws, size_t ws_size,
                              hipStream_t stream) {
  (void)in_sizes; (void)out_size;
  // Assumed flattening: 5 tensors + 257 param leaves in JAX pytree order
  // (dicts sorted by key; ln dicts sort b-before-g; attn sorts k,o,q,v).
  if (n_in < 262) return;
  auto P = [&](int i) -> const float* { return (const float*)d_in[i]; };
  const float* enc   = P(0);                   // (B, HW, D)
  const float* vpos  = P(1);                   // (1, HW, D)
  const float* text  = P(2);                   // (B, T, D)
  const int*   tmask  = (const int*)d_in[3];   // (B, T)
  const int*   tmask2 = (const int*)d_in[4];   // (B, T)

  // ---- workspace layout (bump allocator, float units, 256B aligned)
  float* W = (float*)d_ws;
  size_t off = 0;
  auto alloc = [&](size_t n) -> float* {
    float* p = W + off; off += (n + 63) & ~(size_t)63; return p;
  };
  const int NTOK = Bv * NQ1 * Dv;
  float* bufH    = alloc(NTOK);
  float* bufQ    = alloc(NTOK);
  float* bufX    = alloc(NTOK);
  float* bufQH   = alloc(NTOK);
  float* bufKH   = alloc(NTOK);
  float* bufVH   = alloc(NTOK + 64 * Dv);  // +64 rows: self-AV K padded to 224
  float* bufO    = alloc(NTOK);
  float* bufQPOS = alloc(NTOK);
  float* bufQPRH = alloc((size_t)Bv * Qv * Dv);
  float* bufQPR  = alloc((size_t)Bv * Qv * Dv);
  float* bufQSL  = alloc((size_t)Bv * Qv * Dv);
  float* bufLNQ  = alloc((size_t)Bv * Qv * Dv);
  float* bufB1   = alloc((size_t)Bv * Qv * Dv);
  float* bufB2   = alloc((size_t)Bv * Qv * Dv);
  float* bufDEL  = alloc((size_t)Bv * Qv * 4);
  float* bufREF  = alloc((size_t)Bv * Qv * 4);
  float* bufRE   = alloc((size_t)Bv * Qv * 512);
  float* bufPH   = alloc(Bv * Dv);
  float* bufPHL  = alloc(Bv * Dv);
  float* bufP1   = alloc(Bv * Dv);
  float* bufP2   = alloc(Bv * Dv);
  float* bufPR   = alloc(64);
  float* bufVISK = alloc((size_t)Bv * HWv * Dv);
  float* bufKHV  = alloc((size_t)Bv * HWv * Dv);
  float* bufVHV  = alloc((size_t)Bv * HWv * Dv);
  float* bufRPA  = alloc((size_t)Bv * Qv * 64 * 256);
  float* bufRPB  = alloc((size_t)Bv * Qv * 64 * 256);
  float* bufEX   = alloc((size_t)Bv * Qv * 64 * 8);
  float* bufEY   = alloc((size_t)Bv * Qv * 64 * 8);
  float* bufVB   = alloc((size_t)Bv * HEADSv * NQ1 * HWv);  // bias+scores fused
  float* bufSS   = alloc((size_t)Bv * HEADSv * NQ1 * SSP);  // self scores, padded ld
  float* bufST   = alloc((size_t)Bv * HEADSv * NQ1 * Tv);   // text scores
  float* bufFFN  = alloc((size_t)Bv * NQ1 * INTERv);
  float* bufTF1  = alloc((size_t)Bv * Tv * INTERv);
  float* bufTX   = alloc(Bv * Tv * Dv);
  float* bufTF   = alloc(Bv * Tv * Dv);
  float* bufPOOL = alloc(Bv * Dv);
  float* bufTV   = alloc(Bv * Dv);
  float* bufQP   = alloc((size_t)Bv * Qv * Dv);
  if (off * sizeof(float) > ws_size) return;

  auto dense_l = [&](const float* Xp, int rows, int Kd, int Nd, int wi, int bi,
                     float* Yp, bool relu) {
    launch_gemm(stream, Xp, P(wi), Yp, P(bi), rows, Nd, Kd, Kd, Nd, Nd,
                0, 0, 0, 0, 0, 0, 1, 1, 1.f, false, false, relu);
  };

  // ---- init
  LAUNCH(k_init_h, NTOK, bufH, P(237), P(238));
  LAUNCH(k_init_ref, Bv * Qv * 4, bufREF, P(243));
  LAUNCH(k_add_bcast, Bv * HWv * Dv, bufVISK, enc, vpos, Bv * HWv * Dv, HWv * Dv);
  // zero padded self-score buffer once; pad columns (201..223) stay zero forever
  LAUNCH(k_zero, Bv * HEADSv * NQ1 * SSP, bufSS, Bv * HEADSv * NQ1 * SSP);

  float* outF = (float*)d_out;
  const int OUT_BOX = Bv * Qv * Dv;          // 204800
  const int OUT_LOG = OUT_BOX + Bv * Qv * 4; // 208000
  const int OUT_PRES = OUT_LOG + Bv * Qv;    // 208800

  for (int li = 0; li < Lv; ++li) {
    const int LB = 11 + 36 * li;
    const int SA = LB + 12, TA = LB + 20, VA = LB + 28;  // k.b,k.w,o.b,o.w,q.b,q.w,v.b,v.w

    // ---- query positional embedding
    LAUNCH(k_sine_ref, Bv * Qv * 512, bufREF, bufRE);
    dense_l(bufRE, Bv * Qv, 512, 256, 240, 239, bufQPRH, true);
    dense_l(bufQPRH, Bv * Qv, 256, 256, 242, 241, bufQPR, false);
    LAUNCH(k_qpos, NTOK, bufQPR, bufQPOS);

    // ---- relative position bias -> bufVB
    LAUNCH(k_sine_rpb, Bv * Qv * 64 * 256, bufREF, bufRPA, 0);
    dense_l(bufRPA, Bv * Qv * 64, 256, 256, 245, 244, bufRPB, true);
    dense_l(bufRPB, Bv * Qv * 64, 256, 8, 247, 246, bufEX, false);
    LAUNCH(k_sine_rpb, Bv * Qv * 64 * 256, bufREF, bufRPA, 1);
    dense_l(bufRPA, Bv * Qv * 64, 256, 256, 249, 248, bufRPB, true);
    dense_l(bufRPB, Bv * Qv * 64, 256, 8, 251, 250, bufEY, false);
    LAUNCH(k_bias_expand, Bv * HEADSv * NQ1 * HWv, bufEX, bufEY, bufVB);

    // ---- self attention (scores in padded-ld buffer; AV runs K=224 clean)
    LAUNCH(k_add, NTOK, bufQ, bufH, bufQPOS, NTOK);
    dense_l(bufQ, Bv * NQ1, 256, 256, SA + 5, SA + 4, bufQH, false);
    dense_l(bufQ, Bv * NQ1, 256, 256, SA + 1, SA + 0, bufKH, false);
    dense_l(bufH, Bv * NQ1, 256, 256, SA + 7, SA + 6, bufVH, false);
    launch_gemm(stream, bufQH, bufKH, bufSS, nullptr, NQ1, NQ1, HDv, Dv, Dv, SSP,
                (long long)NQ1 * Dv, (long long)NQ1 * Dv, (long long)HEADSv * NQ1 * SSP,
                HDv, HDv, (long long)NQ1 * SSP, Bv, HEADSv, ATTN_SCALE, true, false, false);
    k_softmax<<<Bv * HEADSv * NQ1, 256, 0, stream>>>(bufSS, nullptr, NQ1, SSP,
                                                     HEADSv * NQ1);
    launch_gemm(stream, bufSS, bufVH, bufO, nullptr, NQ1, HDv, SSP, SSP, Dv, Dv,
                (long long)HEADSv * NQ1 * SSP, (long long)NQ1 * Dv, (long long)NQ1 * Dv,
                (long long)NQ1 * SSP, HDv, HDv, Bv, HEADSv, 1.f, false, false, false);
    dense_l(bufO, Bv * NQ1, 256, 256, SA + 3, SA + 2, bufX, false);
    k_ln<<<(Bv * NQ1 + 7) / 8, 256, 0, stream>>>(bufH, bufX, P(LB + 5), P(LB + 4), bufH, Bv * NQ1);

    // ---- text cross attention
    LAUNCH(k_add, NTOK, bufQ, bufH, bufQPOS, NTOK);
    dense_l(bufQ, Bv * NQ1, 256, 256, TA + 5, TA + 4, bufQH, false);
    dense_l(text, Bv * Tv, 256, 256, TA + 1, TA + 0, bufKH, false);
    dense_l(text, Bv * Tv, 256, 256, TA + 7, TA + 6, bufVH, false);
    launch_gemm(stream, bufQH, bufKH, bufST, nullptr, NQ1, Tv, HDv, Dv, Dv, Tv,
                (long long)NQ1 * Dv, (long long)Tv * Dv, (long long)HEADSv * NQ1 * Tv,
                HDv, HDv, (long long)NQ1 * Tv, Bv, HEADSv, ATTN_SCALE, true, false, false);
    k_softmax<<<Bv * HEADSv * NQ1, 256, 0, stream>>>(bufST, tmask, Tv, Tv, HEADSv * NQ1);
    launch_gemm(stream, bufST, bufVH, bufO, nullptr, NQ1, HDv, Tv, Tv, Dv, Dv,
                (long long)HEADSv * NQ1 * Tv, (long long)Tv * Dv, (long long)NQ1 * Dv,
                (long long)NQ1 * Tv, HDv, HDv, Bv, HEADSv, 1.f, false, false, false);
    dense_l(bufO, Bv * NQ1, 256, 256, TA + 3, TA + 2, bufX, false);
    k_ln<<<(Bv * NQ1 + 7) / 8, 256, 0, stream>>>(bufH, bufX, P(LB + 7), P(LB + 6), bufH, Bv * NQ1);

    // ---- visual cross attention (bias fused via add_c into bufVB)
    LAUNCH(k_add, NTOK, bufQ, bufH, bufQPOS, NTOK);
    dense_l(bufQ, Bv * NQ1, 256, 256, VA + 5, VA + 4, bufQH, false);
    dense_l(bufVISK, Bv * HWv, 256, 256, VA + 1, VA + 0, bufKHV, false);
    dense_l(enc, Bv * HWv, 256, 256, VA + 7, VA + 6, bufVHV, false);
    launch_gemm(stream, bufQH, bufKHV, bufVB, nullptr, NQ1, HWv, HDv, Dv, Dv, HWv,
                (long long)NQ1 * Dv, (long long)HWv * Dv, (long long)HEADSv * NQ1 * HWv,
                HDv, HDv, (long long)NQ1 * HWv, Bv, HEADSv, ATTN_SCALE, true, true, false);
    k_softmax<<<Bv * HEADSv * NQ1, 256, 0, stream>>>(bufVB, nullptr, HWv, HWv,
                                                     HEADSv * NQ1);
    launch_gemm(stream, bufVB, bufVHV, bufO, nullptr, NQ1, HDv, HWv, HWv, Dv, Dv,
                (long long)HEADSv * NQ1 * HWv, (long long)HWv * Dv, (long long)NQ1 * Dv,
                (long long)NQ1 * HWv, HDv, HDv, Bv, HEADSv, 1.f, false, false, false);
    dense_l(bufO, Bv * NQ1, 256, 256, VA + 3, VA + 2, bufX, false);
    k_ln<<<(Bv * NQ1 + 7) / 8, 256, 0, stream>>>(bufH, bufX, P(LB + 9), P(LB + 8), bufH, Bv * NQ1);

    // ---- FFN
    dense_l(bufH, Bv * NQ1, 256, INTERv, LB + 1, LB + 0, bufFFN, true);
    dense_l(bufFFN, Bv * NQ1, INTERv, 256, LB + 3, LB + 2, bufX, false);
    k_ln<<<(Bv * NQ1 + 7) / 8, 256, 0, stream>>>(bufH, bufX, P(LB + 11), P(LB + 10), bufH, Bv * NQ1);

    // ---- box head + reference update
    LAUNCH(k_slice, Bv * Qv * Dv, bufH, bufQSL, Bv, Qv, 1, NQ1, Dv);
    k_ln<<<(Bv * Qv + 7) / 8, 256, 0, stream>>>(bufQSL, nullptr, P(228), P(227), bufLNQ, Bv * Qv);
    dense_l(bufLNQ, Bv * Qv, 256, 256, 6, 5, bufB1, true);
    dense_l(bufB1, Bv * Qv, 256, 256, 8, 7, bufB2, true);
    dense_l(bufB2, Bv * Qv, 256, 4, 10, 9, bufDEL, false);
    LAUNCH(k_update_ref, Bv * Qv * 4, bufREF, bufDEL, Bv * Qv * 4);

    // ---- presence head (clipped logits written straight to output, layer-major)
    LAUNCH(k_slice, Bv * Dv, bufH, bufPH, Bv, 1, 0, NQ1, Dv);
    k_ln<<<1, 256, 0, stream>>>(bufPH, nullptr, P(236), P(235), bufPHL, Bv);
    dense_l(bufPHL, Bv, 256, 256, 230, 229, bufP1, true);
    dense_l(bufP1, Bv, 256, 256, 232, 231, bufP2, true);
    dense_l(bufP2, Bv, 256, 1, 234, 233, bufPR, false);
    k_pres<<<1, 64, 0, stream>>>(bufPR, outF + OUT_PRES + li * Bv);
  }

  // ---- outputs: dec, boxes (= xyxy of final ref), logits
  LAUNCH(k_copy, Bv * Qv * Dv, bufLNQ, outF, Bv * Qv * Dv);
  LAUNCH(k_boxes, Bv * Qv, bufREF, outF + OUT_BOX, Bv * Qv);

  // ---- scoring
  dense_l(text, Bv * Tv, 256, INTERv, 253, 252, bufTF1, true);
  dense_l(bufTF1, Bv * Tv, INTERv, 256, 255, 254, bufTX, false);
  k_ln<<<(Bv * Tv + 7) / 8, 256, 0, stream>>>(bufTX, text, P(257), P(256), bufTF, Bv * Tv);
  LAUNCH(k_pool, Bv * Dv, bufTF, tmask2, bufPOOL);
  dense_l(bufPOOL, Bv, 256, 256, 261, 260, bufTV, false);
  dense_l(bufLNQ, Bv * Qv, 256, 256, 259, 258, bufQP, false);
  k_logits<<<(Bv * Qv * 32 + 255) / 256, 256, 0, stream>>>(bufQP, bufTV, outF + OUT_LOG);
}
